// DecoderBlock_25357486915896
// MI455X (gfx1250) — matrix operand order repaired and checked by reading, not compiled, not run on hardware
//
#include <hip/hip_runtime.h>
#include <stdint.h>

// ---------------------------------------------------------------------------
// DecoderBlock on MI455X (gfx1250): all GEMMs via v_wmma_f32_16x16x32_bf16.
// Wave tile: 32x64 (8 accumulator fragments). No manual double-buffering --
// fragments are loaded into named locals so the compiler batches the loads
// (s_clause) and schedules waits, without spilling accumulators.
// B=64, N=256, DIN=256, H=16, DH=64, d_qkv=6144, d_expand=1024, M=B*N=16384
// ---------------------------------------------------------------------------

typedef __attribute__((ext_vector_type(16))) __bf16 v16bf;
typedef __attribute__((ext_vector_type(8)))  float  v8f;

union FragBF { v16bf v; uint4 u[2]; };

__device__ __forceinline__ unsigned short f2bf(float f) {
  union { float f; unsigned int u; } a; a.f = f;
  unsigned int r = a.u + 0x7FFFu + ((a.u >> 16) & 1u);  // round-to-nearest-even
  return (unsigned short)(r >> 16);
}

// A fragment (16x32 bf16, M x K): lane l16 = row; half=lane/16 selects
// K 0..15 (half=0) vs 16..31 (half=1). Two 16B chunks.
__device__ __forceinline__ v16bf load_a_frag(const unsigned short* rowp, int k0, int half) {
  FragBF f;
  const uint4* p = reinterpret_cast<const uint4*>(rowp + k0 + half * 16);
  f.u[0] = p[0];   // K k0+h*16 .. +7
  f.u[1] = p[1];   // K k0+h*16+8 .. +15
  return f.v;
}
// B fragment (32x16 bf16, K x N) read from Bt[N,K] row-major: lane l16 = column,
// half selects K 0..15 vs 16..31 -> one contiguous 32B run.
__device__ __forceinline__ v16bf load_b_frag(const unsigned short* rowp, int k0, int half) {
  FragBF f;
  const uint4* p = reinterpret_cast<const uint4*>(rowp + k0 + half * 16);
  f.u[0] = p[0];
  f.u[1] = p[1];
  return f.v;
}

// Core: one wave computes a 32 x 64 output tile. K multiple of 32.
template<int K>
__device__ __forceinline__ void wmma_32x64(const unsigned short* __restrict__ arow0,
                                           const unsigned short* __restrict__ arow1,
                                           const unsigned short* __restrict__ bbase,
                                           int ldb, int half, v8f acc[2][4]) {
  const unsigned short* b1p = bbase + (size_t)16 * ldb;
  const unsigned short* b2p = bbase + (size_t)32 * ldb;
  const unsigned short* b3p = bbase + (size_t)48 * ldb;
  for (int k0 = 0; k0 < K; k0 += 32) {
    // issue all 12 b128 loads first; compiler clauses them and staggers waits
    v16bf a0 = load_a_frag(arow0, k0, half);
    v16bf a1 = load_a_frag(arow1, k0, half);
    v16bf b0 = load_b_frag(bbase, k0, half);
    v16bf b1 = load_b_frag(b1p,   k0, half);
    v16bf b2 = load_b_frag(b2p,   k0, half);
    v16bf b3 = load_b_frag(b3p,   k0, half);
    acc[0][0] = __builtin_amdgcn_wmma_f32_16x16x32_bf16(false, a0, false, b0, (short)0, acc[0][0], false, false);
    acc[0][1] = __builtin_amdgcn_wmma_f32_16x16x32_bf16(false, a0, false, b1, (short)0, acc[0][1], false, false);
    acc[0][2] = __builtin_amdgcn_wmma_f32_16x16x32_bf16(false, a0, false, b2, (short)0, acc[0][2], false, false);
    acc[0][3] = __builtin_amdgcn_wmma_f32_16x16x32_bf16(false, a0, false, b3, (short)0, acc[0][3], false, false);
    acc[1][0] = __builtin_amdgcn_wmma_f32_16x16x32_bf16(false, a1, false, b0, (short)0, acc[1][0], false, false);
    acc[1][1] = __builtin_amdgcn_wmma_f32_16x16x32_bf16(false, a1, false, b1, (short)0, acc[1][1], false, false);
    acc[1][2] = __builtin_amdgcn_wmma_f32_16x16x32_bf16(false, a1, false, b2, (short)0, acc[1][2], false, false);
    acc[1][3] = __builtin_amdgcn_wmma_f32_16x16x32_bf16(false, a1, false, b3, (short)0, acc[1][3], false, false);
  }
}

// ---------------------------------------------------------------------------
// Weight transpose + fp32 -> bf16:  W[K,N] -> Wt[N,K]
// ---------------------------------------------------------------------------
__global__ void tconv_kernel(const float* __restrict__ W, unsigned short* __restrict__ Wt,
                             int K, int N) {
  int idx = blockIdx.x * blockDim.x + threadIdx.x;
  if (idx >= K * N) return;
  int k = idx / N, n = idx - k * N;
  Wt[(size_t)n * K + k] = f2bf(W[idx]);
}

// ---------------------------------------------------------------------------
// LayerNorm (one wave per row) with optional pos-enc add and swish, out bf16
// ---------------------------------------------------------------------------
template<int D, bool PE, bool SWISH>
__global__ void ln_kernel(const float* __restrict__ x, const float* __restrict__ w,
                          const float* __restrict__ bvec, unsigned short* __restrict__ out) {
  int wave = (blockIdx.x * blockDim.x + threadIdx.x) >> 5;
  int lane = threadIdx.x & 31;
  constexpr int E = D / 32;
  const float* row = x + (size_t)wave * D;
  float vals[E];
  float s = 0.f;
#pragma unroll
  for (int j = 0; j < E; ++j) { vals[j] = row[j * 32 + lane]; s += vals[j]; }
  for (int off = 16; off; off >>= 1) s += __shfl_xor(s, off, 32);
  float mean = s * (1.f / (float)D);
  float vs = 0.f;
#pragma unroll
  for (int j = 0; j < E; ++j) { float d = vals[j] - mean; vs += d * d; }
  for (int off = 16; off; off >>= 1) vs += __shfl_xor(vs, off, 32);
  float rinv = rsqrtf(vs * (1.f / (float)D) + 1e-5f);
  int pos = wave & 255;
#pragma unroll
  for (int j = 0; j < E; ++j) {
    int c = j * 32 + lane;
    float h = (vals[j] - mean) * rinv * w[c] + bvec[c];
    if (PE) {
      int i2 = c & ~1;
      float ang = (float)pos * __expf(-9.210340371976184f * (float)i2 / (float)D);
      h += (c & 1) ? __cosf(ang) : __sinf(ang);
    }
    if (SWISH) h = h / (1.f + __expf(-h));
    out[(size_t)wave * D + c] = f2bf(h);
  }
}

// ---------------------------------------------------------------------------
// qkv = act @ qkv_w + b : epilogue scatters into q[b,h,n,64], k[b,h,n,64],
// and V transposed vt[b,h,c,m] (free transpose for score@V).
// ---------------------------------------------------------------------------
__global__ void gemm_qkv_kernel(const unsigned short* __restrict__ act,
                                const unsigned short* __restrict__ qkvT,
                                const float* __restrict__ qkv_b,
                                unsigned short* __restrict__ qb,
                                unsigned short* __restrict__ kb,
                                unsigned short* __restrict__ vt) {
  int wv = (blockIdx.x * blockDim.x + threadIdx.x) >> 5;
  int lane = threadIdx.x & 31, l16 = lane & 15, half = lane >> 4;
  int nt = wv % 96, mt = wv / 96;                       // 96 N64-tiles, 512 M32-tiles
  const unsigned short* arow0 = act  + (size_t)(mt * 32 + l16) * 256;
  const unsigned short* arow1 = arow0 + (size_t)16 * 256;
  const unsigned short* bbase = qkvT + (size_t)(nt * 64 + l16) * 256;
  v8f z = {0.f,0.f,0.f,0.f,0.f,0.f,0.f,0.f};
  v8f acc[2][4] = {{z, z, z, z}, {z, z, z, z}};
  wmma_32x64<256>(arow0, arow1, bbase, 256, half, acc);
#pragma unroll
  for (int ra = 0; ra < 2; ++ra) {
#pragma unroll
    for (int j = 0; j < 4; ++j) {
      int col = nt * 64 + j * 16 + l16;
      float bias = qkv_b[col];
#pragma unroll
      for (int r = 0; r < 8; ++r) {
        int m = mt * 32 + ra * 16 + r + half * 8;
        int bi = m >> 8, n = m & 255;
        unsigned short val = f2bf(acc[ra][j][r] + bias);
        if (col < 1024) {
          int h = col >> 6, cc = col & 63;
          qb[((size_t)(bi * 16 + h) * 256 + n) * 64 + cc] = val;
        } else if (col < 2048) {
          int c2 = col - 1024, h = c2 >> 6, cc = c2 & 63;
          kb[((size_t)(bi * 16 + h) * 256 + n) * 64 + cc] = val;
        } else {
          int c2 = col - 2048, h = c2 >> 8, cc = c2 & 255;
          vt[((size_t)(bi * 16 + h) * 256 + cc) * 256 + n] = val;  // transposed V
        }
      }
    }
  }
}

// score[b,h,n,m] = (q[b,h,n,:] . k[b,h,m,:]) * scale^-0.5   (K = 64)
__global__ void gemm_score_kernel(const unsigned short* __restrict__ qb,
                                  const unsigned short* __restrict__ kb,
                                  const float* __restrict__ scale,
                                  unsigned short* __restrict__ score) {
  int wv = (blockIdx.x * blockDim.x + threadIdx.x) >> 5;
  int lane = threadIdx.x & 31, l16 = lane & 15, half = lane >> 4;
  int nt = wv & 3, mt = (wv >> 2) & 7, bh = wv >> 5;    // 4 N64-tiles, 8 M32-tiles
  const unsigned short* arow0 = qb + ((size_t)bh * 256 + mt * 32 + l16) * 64;
  const unsigned short* arow1 = arow0 + (size_t)16 * 64;
  const unsigned short* bbase = kb + ((size_t)bh * 256 + nt * 64 + l16) * 64;
  v8f z = {0.f,0.f,0.f,0.f,0.f,0.f,0.f,0.f};
  v8f acc[2][4] = {{z, z, z, z}, {z, z, z, z}};
  wmma_32x64<64>(arow0, arow1, bbase, 64, half, acc);
  float rs = rsqrtf(scale[0]);
#pragma unroll
  for (int ra = 0; ra < 2; ++ra) {
#pragma unroll
    for (int j = 0; j < 4; ++j) {
      int cm = nt * 64 + j * 16 + l16;
#pragma unroll
      for (int r = 0; r < 8; ++r) {
        int rn = mt * 32 + ra * 16 + r + half * 8;
        score[((size_t)bh * 256 + rn) * 256 + cm] = f2bf(acc[ra][j][r] * rs);
      }
    }
  }
}

// out[b,n,h,c] = sum_m score[b,h,n,m] * v[b,m,h,c]  via vt[b,h,c,m]  (K = 256)
__global__ void gemm_attnout_kernel(const unsigned short* __restrict__ score,
                                    const unsigned short* __restrict__ vt,
                                    unsigned short* __restrict__ outb) {
  int wv = (blockIdx.x * blockDim.x + threadIdx.x) >> 5;
  int lane = threadIdx.x & 31, l16 = lane & 15, half = lane >> 4;
  int nt = wv & 3, mt = (wv >> 2) & 7, bh = wv >> 5;
  int bi = bh >> 4, h = bh & 15;
  const unsigned short* arow0 = score + ((size_t)bh * 256 + mt * 32 + l16) * 256;
  const unsigned short* arow1 = arow0 + (size_t)16 * 256;
  const unsigned short* bbase = vt + ((size_t)bh * 256 + nt * 64 + l16) * 256;
  v8f z = {0.f,0.f,0.f,0.f,0.f,0.f,0.f,0.f};
  v8f acc[2][4] = {{z, z, z, z}, {z, z, z, z}};
  wmma_32x64<256>(arow0, arow1, bbase, 256, half, acc);
#pragma unroll
  for (int ra = 0; ra < 2; ++ra) {
#pragma unroll
    for (int j = 0; j < 4; ++j) {
      int feat = h * 256 + nt * 64 + j * 16 + l16;
#pragma unroll
      for (int r = 0; r < 8; ++r) {
        int token = bi * 256 + mt * 32 + ra * 16 + r + half * 8;
        outb[(size_t)token * 4096 + feat] = f2bf(acc[ra][j][r]);
      }
    }
  }
}

// x2 = x + outb @ merge_w + merge_b   (K = 4096, fp32 out)
__global__ void gemm_merge_kernel(const unsigned short* __restrict__ outb,
                                  const unsigned short* __restrict__ mergeT,
                                  const float* __restrict__ merge_b,
                                  const float* __restrict__ x,
                                  float* __restrict__ x2) {
  int wv = (blockIdx.x * blockDim.x + threadIdx.x) >> 5;
  int lane = threadIdx.x & 31, l16 = lane & 15, half = lane >> 4;
  int nt = wv & 3, mt = wv >> 2;                        // 4 N64-tiles, 512 M32-tiles
  const unsigned short* arow0 = outb + (size_t)(mt * 32 + l16) * 4096;
  const unsigned short* arow1 = arow0 + (size_t)16 * 4096;
  const unsigned short* bbase = mergeT + (size_t)(nt * 64 + l16) * 4096;
  v8f z = {0.f,0.f,0.f,0.f,0.f,0.f,0.f,0.f};
  v8f acc[2][4] = {{z, z, z, z}, {z, z, z, z}};
  wmma_32x64<4096>(arow0, arow1, bbase, 4096, half, acc);
#pragma unroll
  for (int ra = 0; ra < 2; ++ra) {
#pragma unroll
    for (int j = 0; j < 4; ++j) {
      int col = nt * 64 + j * 16 + l16;
      float bias = merge_b[col];
#pragma unroll
      for (int r = 0; r < 8; ++r) {
        int m = mt * 32 + ra * 16 + r + half * 8;
        x2[(size_t)m * 256 + col] = acc[ra][j][r] + bias + x[(size_t)m * 256 + col];
      }
    }
  }
}

// ff = a2 @ ff1_w + ff1_b   (K = 256, N = 1024, fp32 out)
__global__ void gemm_ff1_kernel(const unsigned short* __restrict__ a2,
                                const unsigned short* __restrict__ ff1T,
                                const float* __restrict__ ff1_b,
                                float* __restrict__ ff) {
  int wv = (blockIdx.x * blockDim.x + threadIdx.x) >> 5;
  int lane = threadIdx.x & 31, l16 = lane & 15, half = lane >> 4;
  int nt = wv & 15, mt = wv >> 4;                       // 16 N64-tiles, 512 M32-tiles
  const unsigned short* arow0 = a2 + (size_t)(mt * 32 + l16) * 256;
  const unsigned short* arow1 = arow0 + (size_t)16 * 256;
  const unsigned short* bbase = ff1T + (size_t)(nt * 64 + l16) * 256;
  v8f z = {0.f,0.f,0.f,0.f,0.f,0.f,0.f,0.f};
  v8f acc[2][4] = {{z, z, z, z}, {z, z, z, z}};
  wmma_32x64<256>(arow0, arow1, bbase, 256, half, acc);
#pragma unroll
  for (int ra = 0; ra < 2; ++ra) {
#pragma unroll
    for (int j = 0; j < 4; ++j) {
      int col = nt * 64 + j * 16 + l16;
      float bias = ff1_b[col];
#pragma unroll
      for (int r = 0; r < 8; ++r) {
        int m = mt * 32 + ra * 16 + r + half * 8;
        ff[(size_t)m * 1024 + col] = acc[ra][j][r] + bias;
      }
    }
  }
}

// out = a3 @ ff2_w + ff2_b + x2   (K = 1024, final fp32 output)
__global__ void gemm_ff2_kernel(const unsigned short* __restrict__ a3,
                                const unsigned short* __restrict__ ff2T,
                                const float* __restrict__ ff2_b,
                                const float* __restrict__ x2,
                                float* __restrict__ out) {
  int wv = (blockIdx.x * blockDim.x + threadIdx.x) >> 5;
  int lane = threadIdx.x & 31, l16 = lane & 15, half = lane >> 4;
  int nt = wv & 3, mt = wv >> 2;
  const unsigned short* arow0 = a3 + (size_t)(mt * 32 + l16) * 1024;
  const unsigned short* arow1 = arow0 + (size_t)16 * 1024;
  const unsigned short* bbase = ff2T + (size_t)(nt * 64 + l16) * 1024;
  v8f z = {0.f,0.f,0.f,0.f,0.f,0.f,0.f,0.f};
  v8f acc[2][4] = {{z, z, z, z}, {z, z, z, z}};
  wmma_32x64<1024>(arow0, arow1, bbase, 1024, half, acc);
#pragma unroll
  for (int ra = 0; ra < 2; ++ra) {
#pragma unroll
    for (int j = 0; j < 4; ++j) {
      int col = nt * 64 + j * 16 + l16;
      float bias = ff2_b[col];
#pragma unroll
      for (int r = 0; r < 8; ++r) {
        int m = mt * 32 + ra * 16 + r + half * 8;
        out[(size_t)m * 256 + col] = acc[ra][j][r] + bias + x2[(size_t)m * 256 + col];
      }
    }
  }
}

// ---------------------------------------------------------------------------
extern "C" void kernel_launch(void* const* d_in, const int* in_sizes, int n_in,
                              void* d_out, int out_size, void* d_ws, size_t ws_size,
                              hipStream_t stream) {
  (void)in_sizes; (void)n_in; (void)out_size; (void)ws_size;
  const float* x       = (const float*)d_in[0];
  const float* scale   = (const float*)d_in[1];
  const float* ln1_w   = (const float*)d_in[2];
  const float* ln1_b   = (const float*)d_in[3];
  const float* qkv_w   = (const float*)d_in[4];
  const float* qkv_b   = (const float*)d_in[5];
  const float* merge_w = (const float*)d_in[6];
  const float* merge_b = (const float*)d_in[7];
  const float* ln2_w   = (const float*)d_in[8];
  const float* ln2_b   = (const float*)d_in[9];
  const float* ff1_w   = (const float*)d_in[10];
  const float* ff1_b   = (const float*)d_in[11];
  const float* ln3_w   = (const float*)d_in[12];
  const float* ln3_b   = (const float*)d_in[13];
  const float* ff2_w   = (const float*)d_in[14];
  const float* ff2_b   = (const float*)d_in[15];
  float* out = (float*)d_out;
  char* ws = (char*)d_ws;

  // workspace layout (bytes); score-region reused for ff, outb-region for a3
  const size_t VT_OFF     = 0;                          // 64*16*256*256 bf16
  const size_t SCORE_OFF  = VT_OFF     + 134217728;     // 134 MB (later: ff fp32 67MB)
  const size_t OUTB_OFF   = SCORE_OFF  + 134217728;     // 134 MB (later: a3 bf16 33.5MB)
  const size_t QB_OFF     = OUTB_OFF   + 134217728;     // 33.5 MB
  const size_t KB_OFF     = QB_OFF     + 33554432;      // 33.5 MB
  const size_t ACT_OFF    = KB_OFF     + 33554432;      // 8.4 MB
  const size_t A2_OFF     = ACT_OFF    + 8388608;       // 8.4 MB
  const size_t X2_OFF     = A2_OFF     + 8388608;       // 16.8 MB fp32
  const size_t QKVT_OFF   = X2_OFF     + 16777216;      // 3.1 MB
  const size_t MERGET_OFF = QKVT_OFF   + 3145728;       // 2.1 MB
  const size_t FF1T_OFF   = MERGET_OFF + 2097152;       // 0.5 MB
  const size_t FF2T_OFF   = FF1T_OFF   + 524288;        // 0.5 MB

  unsigned short* vt     = (unsigned short*)(ws + VT_OFF);
  unsigned short* score  = (unsigned short*)(ws + SCORE_OFF);
  unsigned short* outb   = (unsigned short*)(ws + OUTB_OFF);
  unsigned short* qb     = (unsigned short*)(ws + QB_OFF);
  unsigned short* kb     = (unsigned short*)(ws + KB_OFF);
  unsigned short* act    = (unsigned short*)(ws + ACT_OFF);
  unsigned short* a2     = (unsigned short*)(ws + A2_OFF);
  float*          x2     = (float*)(ws + X2_OFF);
  unsigned short* qkvT   = (unsigned short*)(ws + QKVT_OFF);
  unsigned short* mergeT = (unsigned short*)(ws + MERGET_OFF);
  unsigned short* ff1T   = (unsigned short*)(ws + FF1T_OFF);
  unsigned short* ff2T   = (unsigned short*)(ws + FF2T_OFF);
  float*          ffbuf  = (float*)(ws + SCORE_OFF);          // alias (score dead)
  unsigned short* a3     = (unsigned short*)(ws + OUTB_OFF);  // alias (outb dead)

  dim3 blk(256);
  // weight transpose + bf16 convert
  tconv_kernel<<<6144, blk, 0, stream>>>(qkv_w,   qkvT,   256,  6144);
  tconv_kernel<<<4096, blk, 0, stream>>>(merge_w, mergeT, 4096, 256);
  tconv_kernel<<<1024, blk, 0, stream>>>(ff1_w,   ff1T,   256,  1024);
  tconv_kernel<<<1024, blk, 0, stream>>>(ff2_w,   ff2T,   1024, 256);
  // ln1 + pos_enc + swish -> act(bf16)
  ln_kernel<256, true, true><<<2048, blk, 0, stream>>>(x, ln1_w, ln1_b, act);
  // qkv GEMM, scatter to q/k/vT layouts       (16384 x 6144, K=256)
  gemm_qkv_kernel<<<6144, blk, 0, stream>>>(act, qkvT, qkv_b, qb, kb, vt);
  // attention scores                          (per b,h: 256x256, K=64)
  gemm_score_kernel<<<4096, blk, 0, stream>>>(qb, kb, scale, score);
  // score @ V                                 (per b,h: 256x256, K=256)
  gemm_attnout_kernel<<<4096, blk, 0, stream>>>(score, vt, outb);
  // merge projection + residual               (16384 x 256, K=4096)
  gemm_merge_kernel<<<256, blk, 0, stream>>>(outb, mergeT, merge_b, x, x2);
  // ln2 -> a2(bf16)
  ln_kernel<256, false, false><<<2048, blk, 0, stream>>>(x2, ln2_w, ln2_b, a2);
  // ff1                                       (16384 x 1024, K=256)
  gemm_ff1_kernel<<<1024, blk, 0, stream>>>(a2, ff1T, ff1_b, ffbuf);
  // ln3 + swish -> a3(bf16)
  ln_kernel<1024, false, true><<<2048, blk, 0, stream>>>(ffbuf, ln3_w, ln3_b, a3);
  // ff2 + bias + residual -> out (fp32)       (16384 x 256, K=1024)
  gemm_ff2_kernel<<<256, blk, 0, stream>>>(a3, ff2T, ff2_b, x2, out);
}
